// W8A8IntLinear_33097017983072
// MI455X (gfx1250) — 8-bit matrix operand order repaired and checked by reading, compiled, not run on hardware
//
#include <hip/hip_runtime.h>
#include <stdint.h>

#define M_DIM 8192
#define K_DIM 4096
#define N_DIM 4096

#define BM 128
#define BN 128
#define BK 64
#define LDSS 80
#define NKT (K_DIM / BK)

typedef __attribute__((ext_vector_type(8))) int vi8;

#if __has_builtin(__builtin_amdgcn_global_load_async_to_lds_b128)
#define USE_ASYNC 1
#else
#define USE_ASYNC 0
#endif

typedef int v4i_vs __attribute__((__vector_size__(16)));
typedef __attribute__((address_space(1))) v4i_vs g4_t;
typedef __attribute__((address_space(3))) v4i_vs l4_t;

__device__ __forceinline__ void wait_async0() {
#if __has_builtin(__builtin_amdgcn_s_wait_asynccnt)
  __builtin_amdgcn_s_wait_asynccnt(0);
#else
  asm volatile("s_wait_asynccnt 0x0" ::: "memory");
#endif
}

__global__ __launch_bounds__(256) void quant_kernel(const float* __restrict__ x,
                                                    int8_t* __restrict__ q,
                                                    float* __restrict__ scale_a) {
  const int row = blockIdx.x;
  const int t = threadIdx.x;
  const float* xr = x + (size_t)row * K_DIM + t * 16;

  float4 v[4];
#pragma unroll
  for (int i = 0; i < 4; ++i) v[i] = *reinterpret_cast<const float4*>(xr + i * 4);

  float am = 0.0f;
#pragma unroll
  for (int i = 0; i < 4; ++i) {
    am = fmaxf(am, fabsf(v[i].x));
    am = fmaxf(am, fabsf(v[i].y));
    am = fmaxf(am, fabsf(v[i].z));
    am = fmaxf(am, fabsf(v[i].w));
  }

  __shared__ float red[256];
  red[t] = am;
  __syncthreads();
#pragma unroll
  for (int s = 128; s > 0; s >>= 1) {
    if (t < s) red[t] = fmaxf(red[t], red[t + s]);
    __syncthreads();
  }
  const float absmax = red[0];
  const float inv = (absmax > 0.0f) ? (127.0f / absmax) : 0.0f;
  if (t == 0) scale_a[row] = (absmax > 0.0f) ? (absmax * (1.0f / 127.0f)) : 1.0f;

  int pk[4];
#pragma unroll
  for (int i = 0; i < 4; ++i) {
    const float* p = reinterpret_cast<const float*>(&v[i]);
    int w = 0;
#pragma unroll
    for (int j = 0; j < 4; ++j) {
      int qv = __float2int_rn(p[j] * inv);
      qv = max(-128, min(127, qv));
      w |= (qv & 0xFF) << (8 * j);
    }
    pk[i] = w;
  }
  *reinterpret_cast<int4*>(q + (size_t)row * K_DIM + t * 16) =
      make_int4(pk[0], pk[1], pk[2], pk[3]);
}

__global__ __launch_bounds__(256) void gemm_kernel(const int8_t* __restrict__ q,
                                                   const float* __restrict__ scale_a,
                                                   const int8_t* __restrict__ w,
                                                   const float* __restrict__ wscale,
                                                   const float* __restrict__ bias,
                                                   float* __restrict__ out) {
  __shared__ __align__(16) int8_t As[2][BM * LDSS];
  __shared__ __align__(16) int8_t Bs[2][BM * LDSS];

  const int t = threadIdx.x;
  const int lane = t & 31;
  const int wv = t >> 5;
  const int wm = wv >> 1;
  const int wn = wv & 1;
  const int hi = lane >> 4;
  const int l16 = lane & 15;

  const int n0 = blockIdx.x * BN;
  const int m0 = blockIdx.y * BM;

  const int r0 = t >> 2;
  const int o0 = (t & 3) * 16;
  const int r1 = (t + 256) >> 2;
  const int o1 = ((t + 256) & 3) * 16;

  const int8_t* aRow0 = q + (size_t)(m0 + r0) * K_DIM + o0;
  const int8_t* aRow1 = q + (size_t)(m0 + r1) * K_DIM + o1;
  const int8_t* bRow0 = w + (size_t)(n0 + r0) * K_DIM + o0;
  const int8_t* bRow1 = w + (size_t)(n0 + r1) * K_DIM + o1;

  int8_t* lA0 = &As[0][r0 * LDSS + o0];
  int8_t* lA1 = &As[0][r1 * LDSS + o1];
  int8_t* lB0 = &Bs[0][r0 * LDSS + o0];
  int8_t* lB1 = &Bs[0][r1 * LDSS + o1];
  const int bufStep = BM * LDSS;

  vi8 acc[2][4];
#pragma unroll
  for (int mi = 0; mi < 2; ++mi)
#pragma unroll
    for (int ni = 0; ni < 4; ++ni)
#pragma unroll
      for (int r = 0; r < 8; ++r) acc[mi][ni][r] = 0;

#if USE_ASYNC
  __builtin_amdgcn_global_load_async_to_lds_b128((g4_t*)aRow0, (l4_t*)lA0, 0, 0);
  __builtin_amdgcn_global_load_async_to_lds_b128((g4_t*)aRow1, (l4_t*)lA1, 0, 0);
  __builtin_amdgcn_global_load_async_to_lds_b128((g4_t*)bRow0, (l4_t*)lB0, 0, 0);
  __builtin_amdgcn_global_load_async_to_lds_b128((g4_t*)bRow1, (l4_t*)lB1, 0, 0);
  wait_async0();
#else
  {
    int4 ga0 = *reinterpret_cast<const int4*>(aRow0);
    int4 ga1 = *reinterpret_cast<const int4*>(aRow1);
    int4 gb0 = *reinterpret_cast<const int4*>(bRow0);
    int4 gb1 = *reinterpret_cast<const int4*>(bRow1);
    *reinterpret_cast<int4*>(lA0) = ga0;
    *reinterpret_cast<int4*>(lA1) = ga1;
    *reinterpret_cast<int4*>(lB0) = gb0;
    *reinterpret_cast<int4*>(lB1) = gb1;
  }
#endif
  __syncthreads();

  for (int kt = 0; kt < NKT; ++kt) {
    const int p = kt & 1;
    const bool more = (kt + 1) < NKT;

#if USE_ASYNC
    if (more) {
      const int kn = (kt + 1) * BK;
      const int np = (1 - p) * bufStep;
      __builtin_amdgcn_global_load_async_to_lds_b128((g4_t*)(aRow0 + kn), (l4_t*)(lA0 + np), 0, 0);
      __builtin_amdgcn_global_load_async_to_lds_b128((g4_t*)(aRow1 + kn), (l4_t*)(lA1 + np), 0, 0);
      __builtin_amdgcn_global_load_async_to_lds_b128((g4_t*)(bRow0 + kn), (l4_t*)(lB0 + np), 0, 0);
      __builtin_amdgcn_global_load_async_to_lds_b128((g4_t*)(bRow1 + kn), (l4_t*)(lB1 + np), 0, 0);
    }
#else
    int4 ga0, ga1, gb0, gb1;
    if (more) {
      const int kn = (kt + 1) * BK;
      ga0 = *reinterpret_cast<const int4*>(aRow0 + kn);
      ga1 = *reinterpret_cast<const int4*>(aRow1 + kn);
      gb0 = *reinterpret_cast<const int4*>(bRow0 + kn);
      gb1 = *reinterpret_cast<const int4*>(bRow1 + kn);
    }
#endif

    vi8 a[2];
#pragma unroll
    for (int mi = 0; mi < 2; ++mi) {
      const int row = wm * 32 + mi * 16 + l16;
#pragma unroll
      for (int j = 0; j < 4; ++j) {
        int2 d = *reinterpret_cast<const int2*>(&As[p][row * LDSS + (j >> 1) * 32 + hi * 16 + (j & 1) * 8]);
        a[mi][2 * j] = d.x;
        a[mi][2 * j + 1] = d.y;
      }
    }

    vi8 b[4];
#pragma unroll
    for (int ni = 0; ni < 4; ++ni) {
      const int rb = wn * 64 + ni * 16 + l16;
      int4 lo = *reinterpret_cast<const int4*>(&Bs[p][rb * LDSS + hi * 16]);
      int4 hi4 = *reinterpret_cast<const int4*>(&Bs[p][rb * LDSS + 32 + hi * 16]);
      b[ni][0] = lo.x;  b[ni][1] = lo.y;  b[ni][2] = lo.z;  b[ni][3] = lo.w;
      b[ni][4] = hi4.x; b[ni][5] = hi4.y; b[ni][6] = hi4.z; b[ni][7] = hi4.w;
    }

#pragma unroll
    for (int mi = 0; mi < 2; ++mi)
#pragma unroll
      for (int ni = 0; ni < 4; ++ni)
        acc[mi][ni] = __builtin_amdgcn_wmma_i32_16x16x64_iu8(
             true, a[mi],  true, b[ni], acc[mi][ni],
             false,  false);

    if (more) {
#if USE_ASYNC
      wait_async0();
#else
      const int np = (1 - p) * bufStep;
      *reinterpret_cast<int4*>(lA0 + np) = ga0;
      *reinterpret_cast<int4*>(lA1 + np) = ga1;
      *reinterpret_cast<int4*>(lB0 + np) = gb0;
      *reinterpret_cast<int4*>(lB1 + np) = gb1;
#endif
      __syncthreads();
    }
  }

#pragma unroll
  for (int mi = 0; mi < 2; ++mi) {
    const int mb = m0 + wm * 32 + mi * 16 + hi * 8;
    float sa[8];
#pragma unroll
    for (int r = 0; r < 8; ++r) sa[r] = scale_a[mb + r];
#pragma unroll
    for (int ni = 0; ni < 4; ++ni) {
      const int nn = n0 + wn * 64 + ni * 16 + l16;
      const float wsc = wscale[nn];
      const float bs = bias[nn];
#pragma unroll
      for (int r = 0; r < 8; ++r) {
        out[(size_t)(mb + r) * N_DIM + nn] =
            (float)acc[mi][ni][r] * sa[r] * wsc + bs;
      }
    }
  }
}

extern "C" void kernel_launch(void* const* d_in, const int* in_sizes, int n_in,
                              void* d_out, int out_size, void* d_ws, size_t ws_size,
                              hipStream_t stream) {
  const float* x = (const float*)d_in[0];
  const int8_t* wgt = (const int8_t*)d_in[1];
  const float* wsc = (const float*)d_in[2];
  const float* bias = (const float*)d_in[3];
  float* out = (float*)d_out;

  int8_t* q = (int8_t*)d_ws;
  float* scale_a = (float*)((char*)d_ws + (size_t)M_DIM * K_DIM);

  quant_kernel<<<M_DIM, 256, 0, stream>>>(x, q, scale_a);

  dim3 grid(N_DIM / BN, M_DIM / BM);
  gemm_kernel<<<grid, 256, 0, stream>>>(q, scale_a, wgt, wsc, bias, out);
}
